// DecoderVectorized_53154515256007
// MI455X (gfx1250) — matrix operand order repaired and checked by reading, not compiled, not run on hardware
//
#include <hip/hip_runtime.h>
#include <hip/hip_bf16.h>
#include <math.h>

// ---------------------------------------------------------------------------
// CDNA5 (gfx1250, wave32) implementation.
// One wave computes 16x32 (GEMM) / 16x16 (attention) output tiles via
// v_wmma_f32_16x16x32_f16.  Fragment layouts used here (A/B k contiguous, not ISA table order):
//   A (16x32 f16): lane m=lane%16, half=lane/16; vgpr i -> K=half*16+2i(,+1)
//   B (32x16 f16): lane n=lane%16, half=lane/16; vgpr i -> K=half*16+2i(,+1)
//   C/D (16x16 f32): lane n=lane%16; vgpr r -> M=r+8*(lane/16)
// ---------------------------------------------------------------------------

typedef __attribute__((ext_vector_type(16))) _Float16 v16h;
typedef __attribute__((ext_vector_type(8)))  float    v8f;

#define LANE_ (threadIdx.x & 31)

static __device__ __forceinline__ int frag_k_a(int i, int half) {
  return half * 16 + 2 * i;
}

// A fragment, full 32-wide K tile, vectorized float2 loads (k pairs adjacent,
// 8B-aligned: all leading dims are even).
static __device__ __forceinline__ v16h load_a_full(const float* __restrict__ A,
                                                   long ld) {
  const int lane = LANE_, m = lane & 15, half = lane >> 4;
  const float* row = A + (long)m * ld;
  v16h a;
#pragma unroll
  for (int i = 0; i < 8; ++i) {
    int k = frag_k_a(i, half);
    float2 x = *(const float2*)(row + k);
    a[2 * i] = (_Float16)x.x;
    a[2 * i + 1] = (_Float16)x.y;
  }
  return a;
}

// A fragment with zero padding for k >= kmax (dh=24 padded to K=32).
static __device__ __forceinline__ v16h load_a_pad(const float* __restrict__ A,
                                                  long ld, int kmax) {
  const int lane = LANE_, m = lane & 15, half = lane >> 4;
  const float* row = A + (long)m * ld;
  v16h a;
#pragma unroll
  for (int i = 0; i < 8; ++i) {
    int k = frag_k_a(i, half);
    float x0 = (k < kmax) ? row[k] : 0.0f;
    float x1 = (k + 1 < kmax) ? row[k + 1] : 0.0f;
    a[2 * i] = (_Float16)x0;
    a[2 * i + 1] = (_Float16)x1;
  }
  return a;
}

// B fragment from f32 source, element (k,n) = B[min(k,kclamp)*kld + n].
static __device__ __forceinline__ v16h load_b_f32(const float* __restrict__ B,
                                                  long kld, int kclamp) {
  const int lane = LANE_, n = lane & 15, half = lane >> 4;
  v16h b;
#pragma unroll
  for (int i = 0; i < 8; ++i) {
    int k0 = half * 16 + 2 * i;
    int ka = (k0 < kclamp) ? k0 : kclamp;
    int kb = (k0 + 1 < kclamp) ? (k0 + 1) : kclamp;
    b[2 * i]     = (_Float16)B[(long)ka * kld + n];
    b[2 * i + 1] = (_Float16)B[(long)kb * kld + n];
  }
  return b;
}

// B fragment for transposed operand (K^T): element (k,n) = Kp[n*nld + k],
// k >= kmax -> 0.  Used for Q@K^T scores.
static __device__ __forceinline__ v16h load_bt_f32(const float* __restrict__ Kp,
                                                   long nld, int kmax) {
  const int lane = LANE_, n = lane & 15, half = lane >> 4;
  const float* col = Kp + (long)n * nld;
  v16h b;
#pragma unroll
  for (int i = 0; i < 8; ++i) {
    int k = half * 16 + 2 * i;
    float x0 = (k < kmax) ? col[k] : 0.0f;
    float x1 = (k + 1 < kmax) ? col[k + 1] : 0.0f;
    b[2 * i] = (_Float16)x0;
    b[2 * i + 1] = (_Float16)x1;
  }
  return b;
}

// A fragment straight from an LDS 16x32 row-major f16 tile (probabilities).
static __device__ __forceinline__ v16h load_a_lds(const _Float16* T) {
  const int lane = LANE_, m = lane & 15, half = lane >> 4;
  v16h a;
#pragma unroll
  for (int i = 0; i < 8; ++i) {
    int k = frag_k_a(i, half);
    a[2 * i]     = T[m * 32 + k];
    a[2 * i + 1] = T[m * 32 + k + 1];
  }
  return a;
}

static __device__ __forceinline__ v8f wmma32(v16h a, v16h b, v8f c) {
  return __builtin_amdgcn_wmma_f32_16x16x32_f16(false, a, false, b,
                                                (short)0, c, false, false);
}

static __device__ __forceinline__ float redmax16(float x) {
#pragma unroll
  for (int m = 1; m < 16; m <<= 1) x = fmaxf(x, __shfl_xor(x, m, 32));
  return x;
}
static __device__ __forceinline__ float redsum16(float x) {
#pragma unroll
  for (int m = 1; m < 16; m <<= 1) x += __shfl_xor(x, m, 32);
  return x;
}

// ---------------------------------------------------------------------------
// Generic WMMA GEMM:  C[M,N] = A[M,K] @ Bw[K,N] + bias (+ res if HAS_RES)
// M%16==0, N%32==0, K%32==0.  One wave per 16x32 tile (one shared A fragment,
// two B fragments / accumulators per k-step), 8 waves per 256-thread block.
// ---------------------------------------------------------------------------
template <bool HAS_RES>
__global__ __launch_bounds__(256) void gemm_wmma_kernel(
    const float* __restrict__ A, const float* __restrict__ Bw,
    const float* __restrict__ bias, const float* __restrict__ res,
    float* __restrict__ C, int M, int N, int K) {
  int wid = (int)((blockIdx.x * blockDim.x + threadIdx.x) >> 5);
  int tiles_n = N >> 5;
  int tiles = (M >> 4) * tiles_n;
  if (wid >= tiles) return;
  int m0 = (wid / tiles_n) * 16;
  int n0 = (wid % tiles_n) * 32;
  v8f acc0 = {}, acc1 = {};
  for (int k0 = 0; k0 < K; k0 += 32) {
    if (k0 + 32 < K)  // stage next weight tile toward the L0/WGP$
      __builtin_prefetch(Bw + (long)(k0 + 32) * N + n0, 0, 0);
    v16h a = load_a_full(A + (long)m0 * K + k0, K);
    v16h b0 = load_b_f32(Bw + (long)k0 * N + n0, N, 31);
    v16h b1 = load_b_f32(Bw + (long)k0 * N + n0 + 16, N, 31);
    acc0 = wmma32(a, b0, acc0);
    acc1 = wmma32(a, b1, acc1);
  }
  const int n = LANE_ & 15, half = LANE_ >> 4;
  float bn0 = bias[n0 + n];
  float bn1 = bias[n0 + 16 + n];
#pragma unroll
  for (int r = 0; r < 8; ++r) {
    long row = m0 + r + 8 * half;
    float v0 = acc0[r] + bn0;
    float v1 = acc1[r] + bn1;
    if (HAS_RES) {
      v0 += res[row * (long)N + n0 + n];
      v1 += res[row * (long)N + n0 + 16 + n];
    }
    C[row * (long)N + n0 + n] = v0;
    C[row * (long)N + n0 + 16 + n] = v1;
  }
}

// ---------------------------------------------------------------------------
// Build cross-attention queries: Q[(b*15+t-1)*256+nq, d] = grid[nq,d]+tpos[t,d]
// ---------------------------------------------------------------------------
__global__ void build_q_kernel(const float* __restrict__ qg,
                               const float* __restrict__ tp,
                               float* __restrict__ Q) {
  long idx = (long)blockIdx.x * blockDim.x + threadIdx.x;
  if (idx >= 7680L * 192) return;
  long row = idx / 192;
  int d = (int)(idx % 192);
  int nq = (int)(row % 256);
  int t = (int)((row / 256) % 15) + 1;
  Q[idx] = qg[nq * 192 + d] + tp[(long)t * 192 + d];
}

// ---------------------------------------------------------------------------
// LayerNorm over last dim (192). One wave per row.
// ---------------------------------------------------------------------------
__global__ __launch_bounds__(256) void ln_kernel(const float* __restrict__ X,
                                                 const float* __restrict__ g,
                                                 const float* __restrict__ bb,
                                                 float* __restrict__ Y, int M) {
  int row = (int)((blockIdx.x * blockDim.x + threadIdx.x) >> 5);
  if (row >= M) return;
  const int lane = LANE_;
  const float* x = X + (long)row * 192;
  float v[6], s = 0.f, s2 = 0.f;
#pragma unroll
  for (int i = 0; i < 6; ++i) {
    v[i] = x[lane + 32 * i];
    s += v[i];
    s2 += v[i] * v[i];
  }
#pragma unroll
  for (int m = 1; m < 32; m <<= 1) {
    s += __shfl_xor(s, m, 32);
    s2 += __shfl_xor(s2, m, 32);
  }
  float mean = s * (1.0f / 192.0f);
  float var = s2 * (1.0f / 192.0f) - mean * mean;
  float inv = rsqrtf(var + 1e-5f);
  float* y = Y + (long)row * 192;
#pragma unroll
  for (int i = 0; i < 6; ++i) {
    int c = lane + 32 * i;
    y[c] = (v[i] - mean) * inv * g[c] + bb[c];
  }
}

// ---------------------------------------------------------------------------
// Exact-erf GELU, elementwise.
// ---------------------------------------------------------------------------
__global__ void gelu_kernel(float* __restrict__ H, long n) {
  long i = (long)blockIdx.x * blockDim.x + threadIdx.x;
  if (i >= n) return;
  float x = H[i];
  H[i] = 0.5f * x * (1.0f + erff(x * 0.70710678118654752f));
}

// ---------------------------------------------------------------------------
// Cross-attention (flash style).  One wave per (b, t, head, 16-query tile).
// Keys for frame t: rows [b*2048, b*2048 + t*128) of KH/VH (block-causal mask).
// dh = 24 (padded to WMMA K=32).  Online softmax in C-layout registers.
// ---------------------------------------------------------------------------
__global__ __launch_bounds__(128) void cross_attn_kernel(
    const float* __restrict__ QH, const float* __restrict__ KH,
    const float* __restrict__ VH, float* __restrict__ F) {
  __shared__ _Float16 Ptile[4][16 * 32];
  const int wslot = threadIdx.x >> 5;
  const int wid = (int)(blockIdx.x * (blockDim.x >> 5) + wslot);
  if (wid >= 2 * 15 * 8 * 16) return;
  const int qt = wid & 15;
  const int h = (wid >> 4) & 7;
  const int tm1 = (wid >> 7) % 15;
  const int b = wid / 1920;
  const int t = tm1 + 1;
  const long qrow0 = ((long)(b * 15 + tm1) * 256) + qt * 16;
  const int d0 = h * 24;
  const int lane = LANE_, n = lane & 15, half = lane >> 4;
  const float scale = 0.2041241452319315f; // 1/sqrt(24)

  v16h aQ = load_a_pad(QH + qrow0 * 192 + d0, 192, 24);
  v8f o0 = {}, o1 = {};
  const v8f zacc = {};
  float mr[8], lr[8];
#pragma unroll
  for (int r = 0; r < 8; ++r) { mr[r] = -1e30f; lr[r] = 0.0f; }

  _Float16* P = Ptile[wslot];
  const int nk = t * 128; // multiple of 32, no partial tiles
  for (int key0 = 0; key0 < nk; key0 += 32) {
    const long kr = (long)b * 2048 + key0;
    if (key0 + 32 < nk) { // prefetch next key block (K and V rows)
      __builtin_prefetch(KH + (kr + 32) * 192 + d0 + (long)(lane & 15) * 192, 0, 0);
      __builtin_prefetch(VH + (kr + 32) * 192 + d0 + (long)(lane & 15) * 192, 0, 0);
    }
    v16h bk0 = load_bt_f32(KH + kr * 192 + d0, 192, 24);
    v16h bk1 = load_bt_f32(KH + (kr + 16) * 192 + d0, 192, 24);
    v8f s0 = wmma32(aQ, bk0, zacc);
    v8f s1 = wmma32(aQ, bk1, zacc);
#pragma unroll
    for (int r = 0; r < 8; ++r) {
      float a0 = s0[r] * scale, a1 = s1[r] * scale;
      float rm = redmax16(fmaxf(a0, a1));
      float newm = fmaxf(mr[r], rm);
      float alpha = __expf(mr[r] - newm);
      float p0 = __expf(a0 - newm), p1 = __expf(a1 - newm);
      float rs = redsum16(p0 + p1);
      lr[r] = lr[r] * alpha + rs;
      mr[r] = newm;
      o0[r] *= alpha;
      o1[r] *= alpha;
      const int m = r + 8 * half;
      P[m * 32 + n] = (_Float16)p0;
      P[m * 32 + 16 + n] = (_Float16)p1;
    }
    asm volatile("s_wait_dscnt 0" ::: "memory");
    v16h aP = load_a_lds(P);
    v16h bv0 = load_b_f32(VH + kr * 192 + d0, 192, 31);
    v16h bv1 = load_b_f32(VH + kr * 192 + d0 + 16, 192, 31);
    o0 = wmma32(aP, bv0, o0);
    o1 = wmma32(aP, bv1, o1);
  }
#pragma unroll
  for (int r = 0; r < 8; ++r) {
    float invl = 1.0f / lr[r];
    long row = qrow0 + r + 8 * half;
    F[row * 192 + d0 + n] = o0[r] * invl;
    if (n < 8) F[row * 192 + d0 + 16 + n] = o1[r] * invl;
  }
}

// ---------------------------------------------------------------------------
// Axial self-attention (row or col), 16 positions per sequence, one wave per
// (sequence, head).  mode 0: rows (stride 1); mode 1: cols (stride 16 rows).
// ---------------------------------------------------------------------------
__global__ __launch_bounds__(128) void axial_attn_kernel(
    const float* __restrict__ QH, const float* __restrict__ KH,
    const float* __restrict__ VH, float* __restrict__ F, int mode) {
  __shared__ _Float16 Ptile[4][16 * 32];
  const int wslot = threadIdx.x >> 5;
  const int wid = (int)(blockIdx.x * (blockDim.x >> 5) + wslot);
  if (wid >= 480 * 8) return;
  const int h = wid & 7;
  const int seq = wid >> 3;
  const int br = seq >> 4, ii = seq & 15;
  long base;
  int stride;
  if (mode == 0) { base = (long)seq * 16; stride = 1; }
  else           { base = (long)br * 256 + ii; stride = 16; }
  const int d0 = h * 24;
  const int lane = LANE_, n = lane & 15, half = lane >> 4;
  const long pld = (long)stride * 192; // position stride in floats
  const float scale = 0.2041241452319315f;
  const v8f zacc = {};

  v16h aQ = load_a_pad(QH + base * 192 + d0, pld, 24);
  v16h bk = load_bt_f32(KH + base * 192 + d0, pld, 24);
  v8f s = wmma32(aQ, bk, zacc);

  _Float16* P = Ptile[wslot];
#pragma unroll
  for (int r = 0; r < 8; ++r) {
    float a = s[r] * scale;
    float rm = redmax16(a);
    float p = __expf(a - rm);
    float rs = redsum16(p);
    p /= rs;
    const int m = r + 8 * half;
    P[m * 32 + n] = (_Float16)p;
    P[m * 32 + 16 + n] = (_Float16)0.0f; // keys 16..31 are padding
  }
  asm volatile("s_wait_dscnt 0" ::: "memory");
  v16h aP = load_a_lds(P);
  // V fragments: clamp key index to 15 (pad rows multiply by zero probs)
  v16h bv0 = load_b_f32(VH + base * 192 + d0, pld, 15);
  v16h bv1 = load_b_f32(VH + base * 192 + d0 + 16, pld, 15);
  v8f o0 = wmma32(aP, bv0, zacc);
  v8f o1 = wmma32(aP, bv1, zacc);
#pragma unroll
  for (int r = 0; r < 8; ++r) {
    long row = base + (long)(r + 8 * half) * stride;
    F[row * 192 + d0 + n] = o0[r];
    if (n < 8) F[row * 192 + d0 + 16 + n] = o1[r];
  }
}

// ---------------------------------------------------------------------------
// Host-side orchestration
// ---------------------------------------------------------------------------
static inline void launch_gemm(const float* A, const float* Bw, const float* bias,
                               const float* res, float* C, int M, int N, int K,
                               hipStream_t s) {
  int tiles = (M / 16) * (N / 32);
  int blocks = (tiles + 7) / 8;
  if (res)
    gemm_wmma_kernel<true><<<blocks, 256, 0, s>>>(A, Bw, bias, res, C, M, N, K);
  else
    gemm_wmma_kernel<false><<<blocks, 256, 0, s>>>(A, Bw, bias, res, C, M, N, K);
}

extern "C" void kernel_launch(void* const* d_in, const int* in_sizes, int n_in,
                              void* d_out, int out_size, void* d_ws, size_t ws_size,
                              hipStream_t stream) {
  (void)in_sizes; (void)n_in; (void)out_size; (void)ws_size;
  const float* mem   = (const float*)d_in[0];   // [2,16,128,192] -> [4096,192]
  const float* qgrid = (const float*)d_in[1];   // [256,192]
  const float* tpos  = (const float*)d_in[2];   // [2048,192]
  const float* c_wq = (const float*)d_in[3];  const float* c_bq = (const float*)d_in[4];
  const float* c_wk = (const float*)d_in[5];  const float* c_bk = (const float*)d_in[6];
  const float* c_wv = (const float*)d_in[7];  const float* c_bv = (const float*)d_in[8];
  const float* c_wo = (const float*)d_in[9];  const float* c_bo = (const float*)d_in[10];
  const float* r_wq = (const float*)d_in[11]; const float* r_bq = (const float*)d_in[12];
  const float* r_wk = (const float*)d_in[13]; const float* r_bk = (const float*)d_in[14];
  const float* r_wv = (const float*)d_in[15]; const float* r_bv = (const float*)d_in[16];
  const float* r_wo = (const float*)d_in[17]; const float* r_bo = (const float*)d_in[18];
  const float* l_wq = (const float*)d_in[19]; const float* l_bq = (const float*)d_in[20];
  const float* l_wk = (const float*)d_in[21]; const float* l_bk = (const float*)d_in[22];
  const float* l_wv = (const float*)d_in[23]; const float* l_bv = (const float*)d_in[24];
  const float* l_wo = (const float*)d_in[25]; const float* l_bo = (const float*)d_in[26];
  const float* rn_g = (const float*)d_in[27]; const float* rn_b = (const float*)d_in[28];
  const float* ff_ln_g = (const float*)d_in[29]; const float* ff_ln_b = (const float*)d_in[30];
  const float* ff_w1 = (const float*)d_in[31]; const float* ff_b1 = (const float*)d_in[32];
  const float* ff_w2 = (const float*)d_in[33]; const float* ff_b2 = (const float*)d_in[34];
  const float* head_w = (const float*)d_in[35]; const float* head_b = (const float*)d_in[36];
  float* out = (float*)d_out; // [7680, 256] row-major == [2,15,16,16,256]

  const int R = 7680;          // B*(T-1)*H*W rows
  const int D = 192;
  const size_t BUF = (size_t)R * D + 256; // +pad for head-7 d0+16..31 overreads
  float* ws = (float*)d_ws;
  float* W0 = ws + 0 * BUF;        // Q-input / LN output
  float* W1 = ws + 1 * BUF;        // QH
  float* W2 = ws + 2 * BUF;        // KH
  float* W3 = ws + 3 * BUF;        // VH
  float* W4 = ws + 4 * BUF;        // attention output (pre-Wo)
  float* W5 = ws + 5 * BUF;        // residual state z
  float* WH = ws + 6 * BUF;        // FFN hidden [7680, 768]

  // ---- cross attention ----
  {
    long n = (long)R * D;
    build_q_kernel<<<(int)((n + 255) / 256), 256, 0, stream>>>(qgrid, tpos, W0);
  }
  launch_gemm(W0, c_wq, c_bq, nullptr, W1, R, D, D, stream);
  launch_gemm(mem, c_wk, c_bk, nullptr, W2, 4096, D, D, stream);
  launch_gemm(mem, c_wv, c_bv, nullptr, W3, 4096, D, D, stream);
  cross_attn_kernel<<<(3840 + 3) / 4, 128, 0, stream>>>(W1, W2, W3, W4);
  launch_gemm(W4, c_wo, c_bo, nullptr, W5, R, D, D, stream);

  // ---- row attention ----
  ln_kernel<<<(R + 7) / 8, 256, 0, stream>>>(W5, rn_g, rn_b, W0, R);
  launch_gemm(W0, r_wq, r_bq, nullptr, W1, R, D, D, stream);
  launch_gemm(W0, r_wk, r_bk, nullptr, W2, R, D, D, stream);
  launch_gemm(W0, r_wv, r_bv, nullptr, W3, R, D, D, stream);
  axial_attn_kernel<<<(3840 + 3) / 4, 128, 0, stream>>>(W1, W2, W3, W4, 0);
  launch_gemm(W4, r_wo, r_bo, W5, W5, R, D, D, stream);

  // ---- column attention (shared LN params) ----
  ln_kernel<<<(R + 7) / 8, 256, 0, stream>>>(W5, rn_g, rn_b, W0, R);
  launch_gemm(W0, l_wq, l_bq, nullptr, W1, R, D, D, stream);
  launch_gemm(W0, l_wk, l_bk, nullptr, W2, R, D, D, stream);
  launch_gemm(W0, l_wv, l_bv, nullptr, W3, R, D, D, stream);
  axial_attn_kernel<<<(3840 + 3) / 4, 128, 0, stream>>>(W1, W2, W3, W4, 1);
  launch_gemm(W4, l_wo, l_bo, W5, W5, R, D, D, stream);

  // ---- feed-forward ----
  ln_kernel<<<(R + 7) / 8, 256, 0, stream>>>(W5, ff_ln_g, ff_ln_b, W0, R);
  launch_gemm(W0, ff_w1, ff_b1, nullptr, WH, R, 4 * D, D, stream);
  {
    long n = (long)R * 4 * D;
    gelu_kernel<<<(int)((n + 255) / 256), 256, 0, stream>>>(WH, n);
  }
  launch_gemm(WH, ff_w2, ff_b2, W5, W5, R, D, 4 * D, stream);

  // ---- head -> d_out ----
  launch_gemm(W5, head_w, head_b, nullptr, out, R, 256, D, stream);
}
